// FixedPointLSTMWithHardActivations_52931176955976
// MI455X (gfx1250) — 8-bit matrix operand order repaired and checked by reading, compiled, not run on hardware
//
#include <hip/hip_runtime.h>
#include <stdint.h>


typedef int v8i __attribute__((ext_vector_type(8)));

#define BB 64
#define TT 256
#define II 512
#define HH 1024
#define GG 4096


static __device__ __forceinline__ int clampi(int v, int lo, int hi) {
  return min(max(v, lo), hi);
}

static __device__ __forceinline__ int a_frag_off(int m, int kin) {
  int lane = m + (((kin >> 3) & 1) << 4);
  int byt  = ((kin >> 4) << 3) + (kin & 7);
  return lane * 32 + byt;
}

static __device__ __forceinline__ int b_frag_off(int n, int kin) {
  int lane = n + (((kin >> 3) & 1) << 4);
  int byt  = ((kin >> 4) << 3) + (kin & 7);
  return lane * 32 + byt;
}

static __device__ __forceinline__ void mac_phase(const uint8_t* __restrict__ aLoF,
                                                 const int8_t* __restrict__ aHiF,
                                                 const int8_t* __restrict__ bF,
                                                 int chunks, int lane,
                                                 v8i& accHi, v8i& accLo) {
  const size_t lbase = (size_t)lane * 32;
#pragma unroll 4
  for (int c = 0; c < chunks; ++c) {
    const size_t fo = (size_t)c * 1024 + lbase;
    v8i ah = *(const v8i*)(aHiF + fo);
    v8i al = *(const v8i*)((const int8_t*)aLoF + fo);
    v8i bv = *(const v8i*)(bF + fo);
    accHi = __builtin_amdgcn_wmma_i32_16x16x64_iu8(true,  ah, true, bv, accHi, false, false);
    accLo = __builtin_amdgcn_wmma_i32_16x16x64_iu8(false, al, true, bv, accLo, false, false);
  }
}

static __device__ __forceinline__ int quant_acc(int hi, int lo, int bias) {
  float f = (float)hi * 256.0f + (float)lo;
  f = f * (1.0f / 256.0f) + (float)bias;
  return clampi(__float2int_rn(f), -32768, 32767);
}

static __device__ __forceinline__ int hsig_q(int g) {
  return clampi(__float2int_rn((float)g * (1.0f / 6.0f) + 128.0f), 0, 256);
}


__global__ void swizzle_w8_kernel(const float* __restrict__ w, int8_t* __restrict__ o,
                                  int rows, int K) {
  int i = blockIdx.x * blockDim.x + threadIdx.x;
  if (i >= rows * K) return;
  int g = i / K, k = i % K;
  int q = clampi(__float2int_rn(w[i] * 256.0f), -128, 127);
  size_t tile = (size_t)(g >> 4) * (K >> 6) + (k >> 6);
  o[tile * 1024 + b_frag_off(g & 15, k & 63)] = (int8_t)q;
}

__global__ void swizzle_x_kernel(const float* __restrict__ x,
                                 uint8_t* __restrict__ lo, int8_t* __restrict__ hi) {
  int i = blockIdx.x * blockDim.x + threadIdx.x;
  if (i >= BB * TT * II) return;
  int b = i / (TT * II);
  int r = i - b * (TT * II);
  int t = r / II, k = r - t * II;
  int q = clampi(__float2int_rn(x[i] * 256.0f), -32768, 32767);
  size_t tile = ((size_t)(b >> 4) * TT + t) * (II >> 6) + (k >> 6);
  size_t dst  = tile * 1024 + a_frag_off(b & 15, k & 63);
  lo[dst] = (uint8_t)(q & 255);
  hi[dst] = (int8_t)(q >> 8);
}

__global__ void swizzle_h0_kernel(const float* __restrict__ h0,
                                  uint8_t* __restrict__ lo, int8_t* __restrict__ hi) {
  int i = blockIdx.x * blockDim.x + threadIdx.x;
  if (i >= BB * HH) return;
  int b = i / HH, k = i - b * HH;
  int q = clampi(__float2int_rn(h0[i] * 256.0f), -32768, 32767);
  size_t tile = (size_t)(b >> 4) * (HH >> 6) + (k >> 6);
  size_t dst  = tile * 1024 + a_frag_off(b & 15, k & 63);
  lo[dst] = (uint8_t)(q & 255);
  hi[dst] = (int8_t)(q >> 8);
}

__global__ void quant_i32_kernel(const float* __restrict__ b, int* __restrict__ o, int n) {
  int i = blockIdx.x * blockDim.x + threadIdx.x;
  if (i < n) o[i] = clampi(__float2int_rn(b[i] * 256.0f), -32768, 32767);
}

__global__ void finalize_kernel(const uint8_t* __restrict__ hlo, const int8_t* __restrict__ hhi,
                                const int* __restrict__ ci,
                                float* __restrict__ hn, float* __restrict__ cn, int n) {
  int i = blockIdx.x * blockDim.x + threadIdx.x;
  if (i >= n) return;
  int b = i / HH, k = i - b * HH;
  size_t tile = (size_t)(b >> 4) * (HH >> 6) + (k >> 6);
  size_t src  = tile * 1024 + a_frag_off(b & 15, k & 63);
  int h = ((int)hhi[src] << 8) | (int)hlo[src];
  hn[i] = (float)h * (1.0f / 256.0f);
  cn[i] = (float)ci[i] * (1.0f / 256.0f);
}


__global__ __launch_bounds__(128, 1) void lstm_step_kernel(
    const uint8_t* __restrict__ x_lo, const int8_t* __restrict__ x_hi,
    const int8_t* __restrict__ w8ih,
    const int8_t* __restrict__ w8hh,
    const int* __restrict__ bih, const int* __restrict__ bhh,
    const uint8_t* __restrict__ h_lo_r, const int8_t* __restrict__ h_hi_r,
    uint8_t* __restrict__ h_lo_w, int8_t* __restrict__ h_hi_w,
    int* __restrict__ c_i,
    float* __restrict__ out,
    int t) {
  __shared__ int lds_g[4][16][16];

  const int tid  = threadIdx.x;
  const int wave = tid >> 5;
  const int lane = tid & 31;
  const int lm   = lane & 15;
  const int lh   = lane >> 4;
  const int bt   = blockIdx.x >> 6;
  const int ht   = blockIdx.x & 63;
  const int b0   = bt * 16;
  const int h0   = ht * 16;

  const int grow = wave * HH + h0 + lm;
  const int ntile = wave * 64 + ht;

  const uint8_t* aLoX = x_lo + (((size_t)bt * TT + t) * (II >> 6)) * 1024;
  const int8_t*  aHiX = x_hi + (((size_t)bt * TT + t) * (II >> 6)) * 1024;
  const int8_t*  bX   = w8ih + ((size_t)ntile * (II >> 6)) * 1024;
  v8i accHi = {}; v8i accLo = {};
  mac_phase(aLoX, aHiX, bX, II >> 6, lane, accHi, accLo);
  const int biasX = bih[grow];
  int gx[8];
#pragma unroll
  for (int r = 0; r < 8; ++r) gx[r] = quant_acc(accHi[r], accLo[r], biasX);

  const uint8_t* aLoH = h_lo_r + ((size_t)bt * (HH >> 6)) * 1024;
  const int8_t*  aHiH = h_hi_r + ((size_t)bt * (HH >> 6)) * 1024;
  const int8_t*  bH   = w8hh + ((size_t)ntile * (HH >> 6)) * 1024;
  v8i acc2Hi = {}; v8i acc2Lo = {};
  mac_phase(aLoH, aHiH, bH, HH >> 6, lane, acc2Hi, acc2Lo);
  const int biasH = bhh[grow];
#pragma unroll
  for (int r = 0; r < 8; ++r) {
    int gh = quant_acc(acc2Hi[r], acc2Lo[r], biasH);
    int g  = clampi(gx[r] + gh, -32768, 32767);
    lds_g[wave][r + lh * 8][lm] = g;
  }
  __syncthreads();

#pragma unroll
  for (int j = 0; j < 2; ++j) {
    const int e = tid * 2 + j;
    const int m = e >> 4, n = e & 15;
    const int gI = lds_g[0][m][n];
    const int gF = lds_g[1][m][n];
    const int gG = lds_g[2][m][n];
    const int gO = lds_g[3][m][n];

    const int ii = hsig_q(gI);
    const int ff = hsig_q(gF);
    const int oo = hsig_q(gO);
    const int tg = clampi(gG, -256, 256);

    const size_t idx = (size_t)(b0 + m) * HH + (h0 + n);
    const int cOld = c_i[idx];
    int cNew = __float2int_rn((float)(ff * cOld + ii * tg) * (1.0f / 256.0f));
    cNew = clampi(cNew, -32768, 32767);
    c_i[idx] = cNew;

    const int th = clampi(cNew, -256, 256);
    const int hN = __float2int_rn((float)(oo * th) * (1.0f / 256.0f));

    const int hcol = h0 + n;
    const size_t tile = (size_t)bt * (HH >> 6) + (hcol >> 6);
    const size_t hds  = tile * 1024 + a_frag_off(m, hcol & 63);
    h_lo_w[hds] = (uint8_t)(hN & 255);
    h_hi_w[hds] = (int8_t)(hN >> 8);

    out[((size_t)(b0 + m) * TT + t) * HH + (h0 + n)] = (float)hN * (1.0f / 256.0f);
  }
}


extern "C" void kernel_launch(void* const* d_in, const int* in_sizes, int n_in,
                              void* d_out, int out_size, void* d_ws, size_t ws_size,
                              hipStream_t stream) {
  (void)in_sizes; (void)n_in; (void)out_size; (void)ws_size;
  const float* x    = (const float*)d_in[0];
  const float* w_ih = (const float*)d_in[1];
  const float* w_hh = (const float*)d_in[2];
  const float* b_ih = (const float*)d_in[3];
  const float* b_hh = (const float*)d_in[4];
  const float* h0   = (const float*)d_in[5];
  const float* c0   = (const float*)d_in[6];
  float* out = (float*)d_out;

  uint8_t* base = (uint8_t*)d_ws;
  size_t off = 0;
  auto carve = [&](size_t bytes) -> void* {
    void* r = base + off;
    off = (off + bytes + 255) & ~(size_t)255;
    return r;
  };
  int8_t*  w8ih = (int8_t*)carve((size_t)GG * II);
  int8_t*  w8hh = (int8_t*)carve((size_t)GG * HH);
  int*     bihq = (int*)carve((size_t)GG * 4);
  int*     bhhq = (int*)carve((size_t)GG * 4);
  uint8_t* xlo  = (uint8_t*)carve((size_t)BB * TT * II);
  int8_t*  xhi  = (int8_t*)carve((size_t)BB * TT * II);
  uint8_t* hlo[2]; int8_t* hhi[2];
  hlo[0] = (uint8_t*)carve((size_t)BB * HH);
  hhi[0] = (int8_t*)carve((size_t)BB * HH);
  hlo[1] = (uint8_t*)carve((size_t)BB * HH);
  hhi[1] = (int8_t*)carve((size_t)BB * HH);
  int*     ci   = (int*)carve((size_t)BB * HH * 4);

  const int TPB = 256;
  auto blocks = [&](int n) { return (n + TPB - 1) / TPB; };

  swizzle_w8_kernel<<<blocks(GG * II), TPB, 0, stream>>>(w_ih, w8ih, GG, II);
  swizzle_w8_kernel<<<blocks(GG * HH), TPB, 0, stream>>>(w_hh, w8hh, GG, HH);
  quant_i32_kernel<<<blocks(GG), TPB, 0, stream>>>(b_ih, bihq, GG);
  quant_i32_kernel<<<blocks(GG), TPB, 0, stream>>>(b_hh, bhhq, GG);
  swizzle_x_kernel<<<blocks(BB * TT * II), TPB, 0, stream>>>(x, xlo, xhi);
  swizzle_h0_kernel<<<blocks(BB * HH), TPB, 0, stream>>>(h0, hlo[0], hhi[0]);
  quant_i32_kernel<<<blocks(BB * HH), TPB, 0, stream>>>(c0, ci, BB * HH);

  for (int t = 0; t < TT; ++t) {
    const int rb = t & 1, wb = (t + 1) & 1;
    lstm_step_kernel<<<256, 128, 0, stream>>>(
        xlo, xhi, w8ih, w8hh, bihq, bhhq,
        hlo[rb], hhi[rb], hlo[wb], hhi[wb],
        ci, out, t);
  }

  finalize_kernel<<<blocks(BB * HH), TPB, 0, stream>>>(
      hlo[0], hhi[0], ci,
      out + (size_t)BB * TT * HH,
      out + (size_t)BB * TT * HH + (size_t)BB * HH,
      BB * HH);
}
